// DIEN_62285615727311
// MI455X (gfx1250) — compile-verified, matrix operand order checked by reading, not run on hardware
//
#include <hip/hip_runtime.h>
#include <hip/hip_bf16.h>
#include <math.h>

#define BB 2048
#define TT 100
#define DD 128
#define HH 128

typedef __bf16 bf16;
typedef __attribute__((ext_vector_type(16))) __bf16 bf16x16;
typedef __attribute__((ext_vector_type(8)))  __bf16 bf16x8;
typedef __attribute__((ext_vector_type(8)))  float  f32x8;

__device__ __forceinline__ f32x8 wmma_bf(bf16x16 a, bf16x16 b, f32x8 c) {
    // D = A(16x32) * B(32x16) + C, f32 accumulate
    return __builtin_amdgcn_wmma_f32_16x16x32_bf16(false, a, false, b, (short)0, c, false, false);
}

__device__ __forceinline__ bf16x8 cvt8(float4 a, float4 b) {
    bf16x8 o;
    o[0] = (bf16)a.x; o[1] = (bf16)a.y; o[2] = (bf16)a.z; o[3] = (bf16)a.w;
    o[4] = (bf16)b.x; o[5] = (bf16)b.y; o[6] = (bf16)b.z; o[7] = (bf16)b.w;
    return o;
}

__device__ __forceinline__ bf16x8 zero8() {
    bf16x8 o;
#pragma unroll
    for (int i = 0; i < 8; ++i) o[i] = (bf16)0.f;
    return o;
}

// Fragment loader, CDNA5 16-bit operand layout: lane&15 -> row, lane>>4 -> K-half.
// Per lane the 16 elements are two contiguous 8-element (16B) runs: [kh..kh+7] and [16+kh..16+kh+7].
// Requires (ld, k_base) multiples of 8 so the 16B vector loads stay aligned.
__device__ __forceinline__ bf16x16 load_frag(const bf16* s, int row_base, int k_base, int ld) {
    int lane = threadIdx.x & 31;
    const bf16* p = s + (size_t)(row_base + (lane & 15)) * ld + k_base;
    int kh = (lane >> 4) << 3;
    bf16x8 lo = *(const bf16x8*)(p + kh);
    bf16x8 hi = *(const bf16x8*)(p + 16 + kh);
    bf16x16 f;
#pragma unroll
    for (int e = 0; e < 8; ++e) { f[e] = lo[e]; f[e + 8] = hi[e]; }
    return f;
}

__device__ __forceinline__ float sigmf(float x) { return 1.f / (1.f + __expf(-x)); }

// -------- generic NT GEMM: C[M,N] = act(A[M,K] @ W[N,K]^T + bias) --------
// A f32, W bf16, C f32. M multiple of 64, K multiple of 8. grid (M/64, ceil(N/32)), block 256.
__global__ __launch_bounds__(256)
void gemm_nt_kernel(const float* __restrict__ A, const bf16* __restrict__ W,
                    const float* __restrict__ bias, float* __restrict__ C,
                    int M, int N, int K, int act) {
    __shared__ __align__(16) bf16 As[64 * 40];
    __shared__ __align__(16) bf16 Bs[32 * 40];
    const int tid  = threadIdx.x;
    const int gm   = blockIdx.x * 64;
    const int gn   = blockIdx.y * 32;
    const int wave = tid >> 5, lane = tid & 31;
    const int wm = (wave & 3) * 16, wn = (wave >> 2) * 16;
    f32x8 acc = {0.f,0.f,0.f,0.f,0.f,0.f,0.f,0.f};
    for (int k0 = 0; k0 < K; k0 += 32) {
        {   // A tile 64x32: vector load f32 -> packed bf16, one 16B LDS store
            int r  = tid >> 2;
            int kq = (tid & 3) * 8;
            bf16x8 pack;
            if (k0 + kq + 8 <= K) {
                const float* ap = A + (size_t)(gm + r) * K + k0 + kq;
                pack = cvt8(*(const float4*)ap, *(const float4*)(ap + 4));
            } else {
                pack = zero8();
            }
            *(bf16x8*)(&As[r * 40 + kq]) = pack;
        }
        if (tid < 128) { // B tile 32x32: one 16B global load, one 16B LDS store
            int r  = tid >> 2;
            int kq = (tid & 3) * 8;
            int n  = gn + r;
            bf16x8 pack = (n < N && (k0 + kq + 8 <= K))
                        ? *(const bf16x8*)(W + (size_t)n * K + k0 + kq)
                        : zero8();
            *(bf16x8*)(&Bs[r * 40 + kq]) = pack;
        }
        __syncthreads();
        bf16x16 af = load_frag(As, wm, 0, 40);
        bf16x16 bw = load_frag(Bs, wn, 0, 40);
        acc = wmma_bf(af, bw, acc);
        __syncthreads();
    }
    int n  = gn + wn + (lane & 15);
    int mb = (lane >> 4) * 8;
    if (n < N) {
        float bv = bias ? bias[n] : 0.f;
#pragma unroll
        for (int r = 0; r < 8; ++r) {
            float v = acc[r] + bv;
            if (act == 1) v = fmaxf(v, 0.f);
            else if (act == 2) v = sigmf(v);
            C[(size_t)(gm + wm + mb + r) * N + n] = v;
        }
    }
}

// -------- attention layer0: h1 = relu([k,q,k-q,q*k] @ W0^T + b0), K=512, N=80 --------
__global__ __launch_bounds__(256)
void att0_kernel(const float* __restrict__ keys, const float* __restrict__ item,
                 const bf16* __restrict__ W0, const float* __restrict__ b0,
                 float* __restrict__ h1) {
    __shared__ __align__(16) bf16 As[64 * 40];
    const int tid  = threadIdx.x;
    const int gm   = blockIdx.x * 64;
    const int gn   = blockIdx.y * 32;
    const int wave = tid >> 5, lane = tid & 31;
    const int wm = (wave & 3) * 16, wn = (wave >> 2) * 16;
    const bool nvalid = (gn + wn) < 80;
    f32x8 acc = {0.f,0.f,0.f,0.f,0.f,0.f,0.f,0.f};
    for (int k0 = 0; k0 < 512; k0 += 32) {
        int seg = k0 >> 7;     // 0:k 1:q 2:k-q 3:q*k
        int jb  = k0 & 127;
        {
            int r   = tid >> 2;
            int kq  = (tid & 3) * 8;
            int row = gm + r;
            int b   = row / TT;
            const float* kp = keys + (size_t)row * HH + jb + kq;
            const float* qp = item + (size_t)b * DD + jb + kq;
            float4 ka = *(const float4*)kp, kb2 = *(const float4*)(kp + 4);
            float4 qa = *(const float4*)qp, qb2 = *(const float4*)(qp + 4);
            float kv[8] = {ka.x,ka.y,ka.z,ka.w,kb2.x,kb2.y,kb2.z,kb2.w};
            float qv[8] = {qa.x,qa.y,qa.z,qa.w,qb2.x,qb2.y,qb2.z,qb2.w};
            bf16x8 pack;
#pragma unroll
            for (int i = 0; i < 8; ++i) {
                float v = (seg == 0) ? kv[i] : (seg == 1) ? qv[i]
                        : (seg == 2) ? (kv[i] - qv[i]) : (qv[i] * kv[i]);
                pack[i] = (bf16)v;
            }
            *(bf16x8*)(&As[r * 40 + kq]) = pack;
        }
        __syncthreads();
        if (nvalid) {   // wave-uniform guard: EXEC all-1s inside for WMMA
            bf16x16 af = load_frag(As, wm, 0, 40);
            bf16x16 bw = load_frag(W0, gn + wn, k0, 512);
            acc = wmma_bf(af, bw, acc);
        }
        __syncthreads();
    }
    int n  = gn + wn + (lane & 15);
    int mb = (lane >> 4) * 8;
    if (nvalid && n < 80) {
        float bv = b0[n];
#pragma unroll
        for (int r = 0; r < 8; ++r)
            h1[(size_t)(gm + wm + mb + r) * 80 + n] = fmaxf(acc[r] + bv, 0.f);
    }
}

// -------- GRU scan: persistent over t, 16 batch rows per workgroup --------
__global__ __launch_bounds__(256)
void gru_scan_kernel(const float* __restrict__ GI, const bf16* __restrict__ Whh,
                     const float* __restrict__ bhh, float* __restrict__ keys,
                     bf16* __restrict__ keys_bf) {
    __shared__ __align__(16) float h_s[16 * 132];
    __shared__ __align__(16) bf16  hb [16 * 136];
    __shared__ __align__(16) float gh [16 * 388];
    const int tid  = threadIdx.x;
    const int gb   = blockIdx.x * 16;
    const int wave = tid >> 5, lane = tid & 31;
    const int erow = tid >> 4, ej0 = (tid & 15) * 8;   // contiguous 8-wide element mapping
    for (int e = tid; e < 16 * 128; e += 256)
        h_s[(e >> 7) * 132 + (e & 127)] = 0.f;
    __syncthreads();
    for (int t = 0; t < TT; ++t) {
        {   // h -> bf16 staging, 16B store
            bf16x8 v;
#pragma unroll
            for (int i = 0; i < 8; ++i) v[i] = (bf16)h_s[erow * 132 + ej0 + i];
            *(bf16x8*)(&hb[erow * 136 + ej0]) = v;
        }
        __syncthreads();
        bf16x16 af[4];
#pragma unroll
        for (int kk = 0; kk < 4; ++kk) af[kk] = load_frag(hb, 0, kk * 32, 136);
#pragma unroll
        for (int i = 0; i < 3; ++i) {      // 24 N-tiles over 8 waves
            int nt = wave + 8 * i;
            f32x8 acc = {0.f,0.f,0.f,0.f,0.f,0.f,0.f,0.f};
#pragma unroll
            for (int kk = 0; kk < 4; ++kk)
                acc = wmma_bf(af[kk], load_frag(Whh, nt * 16, kk * 32, 128), acc);
            int n  = nt * 16 + (lane & 15);
            int mb = (lane >> 4) * 8;
#pragma unroll
            for (int r = 0; r < 8; ++r) gh[(mb + r) * 388 + n] = acc[r];
        }
        __syncthreads();
        {   // gates: 8 contiguous j per thread -> vector GI reads / keys writes
            const float* gi = GI + ((size_t)(gb + erow) * TT + t) * 384;
            float hv[8];
#pragma unroll
            for (int i = 0; i < 8; ++i) {
                int j = ej0 + i;
                float hr = gh[erow * 388 + j]       + bhh[j];
                float hz = gh[erow * 388 + 128 + j] + bhh[128 + j];
                float hn = gh[erow * 388 + 256 + j] + bhh[256 + j];
                float rg = sigmf(gi[j] + hr);
                float zg = sigmf(gi[128 + j] + hz);
                float ng = tanhf(gi[256 + j] + rg * hn);
                float hold = h_s[erow * 132 + j];
                hv[i] = (1.f - zg) * ng + zg * hold;
                h_s[erow * 132 + j] = hv[i];
            }
            size_t o = ((size_t)(gb + erow) * TT + t) * 128 + ej0;
            *(float4*)(keys + o)     = make_float4(hv[0], hv[1], hv[2], hv[3]);
            *(float4*)(keys + o + 4) = make_float4(hv[4], hv[5], hv[6], hv[7]);
            bf16x8 kb;
#pragma unroll
            for (int i = 0; i < 8; ++i) kb[i] = (bf16)hv[i];
            *(bf16x8*)(keys_bf + o) = kb;
        }
        __syncthreads();
    }
}

// -------- AUGRU scan --------
__global__ __launch_bounds__(256)
void augru_scan_kernel(const bf16* __restrict__ keys_bf, const bf16* __restrict__ Wru,
                       const bf16* __restrict__ Wh, const float* __restrict__ br,
                       const float* __restrict__ bu, const float* __restrict__ bh,
                       const float* __restrict__ pooled, const int* __restrict__ slen,
                       float* __restrict__ attfeat) {
    __shared__ __align__(16) bf16  ti [16 * 264];
    __shared__ __align__(16) float g1 [16 * 260];
    __shared__ __align__(16) float g2 [16 * 132];
    __shared__ __align__(16) float u_s[16 * 128];
    __shared__ __align__(16) float h_s[16 * 132];
    const int tid  = threadIdx.x;
    const int gb   = blockIdx.x * 16;
    const int wave = tid >> 5, lane = tid & 31;
    const int erow = tid >> 4, ej0 = (tid & 15) * 8;
    for (int e = tid; e < 16 * 128; e += 256)
        h_s[(e >> 7) * 132 + (e & 127)] = 0.f;
    __syncthreads();
    for (int t = 0; t < TT; ++t) {
        {   // ti = [h, x_t], 16 contiguous elements per thread
            int j0 = (tid & 15) * 16;
            if (j0 < 128) {
                bf16x8 v0, v1;
#pragma unroll
                for (int i = 0; i < 8; ++i) {
                    v0[i] = (bf16)h_s[erow * 132 + j0 + i];
                    v1[i] = (bf16)h_s[erow * 132 + j0 + 8 + i];
                }
                *(bf16x8*)(&ti[erow * 264 + j0])     = v0;
                *(bf16x8*)(&ti[erow * 264 + j0 + 8]) = v1;
            } else {
                const bf16* src = keys_bf + ((size_t)(gb + erow) * TT + t) * 128 + (j0 - 128);
                *(bf16x8*)(&ti[erow * 264 + j0])     = *(const bf16x8*)(src);
                *(bf16x8*)(&ti[erow * 264 + j0 + 8]) = *(const bf16x8*)(src + 8);
            }
        }
        __syncthreads();
        bf16x16 af[8];
#pragma unroll
        for (int kk = 0; kk < 8; ++kk) af[kk] = load_frag(ti, 0, kk * 32, 264);
#pragma unroll
        for (int i = 0; i < 2; ++i) {      // G1 = ti @ [Wr;Wu]^T  (N=256)
            int nt = wave + 8 * i;
            f32x8 acc = {0.f,0.f,0.f,0.f,0.f,0.f,0.f,0.f};
#pragma unroll
            for (int kk = 0; kk < 8; ++kk)
                acc = wmma_bf(af[kk], load_frag(Wru, nt * 16, kk * 32, 256), acc);
            int n  = nt * 16 + (lane & 15);
            int mb = (lane >> 4) * 8;
#pragma unroll
            for (int r = 0; r < 8; ++r) g1[(mb + r) * 260 + n] = acc[r];
        }
        __syncthreads();
        {   // r,u ; ti2 = [h*r, x_t]
            bf16x8 hr;
#pragma unroll
            for (int i = 0; i < 8; ++i) {
                int j = ej0 + i;
                float rg = sigmf(g1[erow * 260 + j] + br[j]);
                u_s[erow * 128 + j] = sigmf(g1[erow * 260 + 128 + j] + bu[j]);
                hr[i] = (bf16)(h_s[erow * 132 + j] * rg);
            }
            *(bf16x8*)(&ti[erow * 264 + ej0]) = hr;
        }
        __syncthreads();
#pragma unroll
        for (int kk = 0; kk < 8; ++kk) af[kk] = load_frag(ti, 0, kk * 32, 264);
        {   // G2 = [h*r, x] @ Wh^T  (N=128)
            int nt = wave;
            f32x8 acc = {0.f,0.f,0.f,0.f,0.f,0.f,0.f,0.f};
#pragma unroll
            for (int kk = 0; kk < 8; ++kk)
                acc = wmma_bf(af[kk], load_frag(Wh, nt * 16, kk * 32, 256), acc);
            int n  = nt * 16 + (lane & 15);
            int mb = (lane >> 4) * 8;
#pragma unroll
            for (int r = 0; r < 8; ++r) g2[(mb + r) * 132 + n] = acc[r];
        }
        __syncthreads();
        {   // h update + select at t == len-1
            float a = pooled[(size_t)(gb + erow) * 128 + t];   // att_seq[t] = pooled[:, t]
            bool last = (t == slen[gb + erow] - 1);
#pragma unroll
            for (int i = 0; i < 8; ++i) {
                int j = ej0 + i;
                float hh = tanhf(g2[erow * 132 + j] + bh[j]);
                float up = a * u_s[erow * 128 + j];
                float hold = h_s[erow * 132 + j];
                float hnew = (1.f - up) * hold + up * hh;
                h_s[erow * 132 + j] = hnew;
                if (last) attfeat[(size_t)(gb + erow) * 128 + j] = hnew;
            }
        }
        __syncthreads();
    }
}

// -------- small kernels --------
__global__ void cvt_kernel(const float* __restrict__ s, bf16* __restrict__ d, int n8) {
    int i = blockIdx.x * 256 + threadIdx.x;
    if (i >= n8) return;
    const float* sp = s + (size_t)i * 8;
    *(bf16x8*)(d + (size_t)i * 8) = cvt8(*(const float4*)sp, *(const float4*)(sp + 4));
}

__global__ void hist_sum_kernel(const float* __restrict__ x, const int* __restrict__ mask,
                                const int* __restrict__ slen, float* __restrict__ hist) {
    int b = blockIdx.x, d = threadIdx.x;
    float acc = 0.f;
    for (int t = 0; t < TT; ++t)
        if (mask[b * TT + t]) acc += x[((size_t)b * TT + t) * DD + d];
    hist[(size_t)b * DD + d] = acc / (float)slen[b];
}

__global__ void score_kernel(const float* __restrict__ h2, const float* __restrict__ W2,
                             const float* __restrict__ b2, float* __restrict__ score, int M) {
    int i = blockIdx.x * 256 + threadIdx.x;
    if (i >= M) return;
    float acc = b2[0];
    const float* hp = h2 + (size_t)i * 40;
#pragma unroll
    for (int k = 0; k < 40; k += 4) {
        float4 v = *(const float4*)(hp + k);
        float4 w = *(const float4*)(W2 + k);
        acc += v.x * w.x + v.y * w.y + v.z * w.z + v.w * w.w;
    }
    score[i] = fmaxf(acc, 0.f);
}

__global__ void softmax_kernel(const float* __restrict__ score, const int* __restrict__ mask,
                               float* __restrict__ attn) {
    __shared__ float red[128];
    int b = blockIdx.x, t = threadIdx.x;
    bool valid = (t < TT) && (mask[b * TT + t] != 0);
    float s = valid ? score[b * TT + t] : -1e30f;
    red[t] = s; __syncthreads();
    for (int o = 64; o > 0; o >>= 1) { if (t < o) red[t] = fmaxf(red[t], red[t + o]); __syncthreads(); }
    float mx = red[0]; __syncthreads();
    float e = valid ? __expf(s - mx) : 0.f;
    red[t] = e; __syncthreads();
    for (int o = 64; o > 0; o >>= 1) { if (t < o) red[t] += red[t + o]; __syncthreads(); }
    float inv = 1.f / red[0];
    if (t < TT) attn[b * TT + t] = e * inv;
}

__global__ void pooled_kernel(const float* __restrict__ attn, const float* __restrict__ keys,
                              float* __restrict__ pooled) {
    __shared__ float sa[TT];
    int b = blockIdx.x, d = threadIdx.x;
    if (d < TT) sa[d] = attn[b * TT + d];
    __syncthreads();
    float acc = 0.f;
    for (int t = 0; t < TT; ++t) acc += sa[t] * keys[((size_t)b * TT + t) * HH + d];
    pooled[(size_t)b * HH + d] = acc;
}

__global__ void comb_kernel(const float* __restrict__ user, const float* __restrict__ item,
                            const float* __restrict__ hist, const float* __restrict__ attf,
                            float* __restrict__ comb) {
    int i = blockIdx.x * 256 + threadIdx.x;
    if (i >= BB * 640) return;
    int b = i / 640, j = i % 640;
    float v;
    if      (j < 128) v = user[b * 128 + j];
    else if (j < 256) v = item[b * 128 + (j - 128)];
    else if (j < 384) v = hist[b * 128 + (j - 256)];
    else if (j < 512) v = item[b * 128 + (j - 384)] * hist[b * 128 + (j - 384)];
    else              v = attf[b * 128 + (j - 512)];
    comb[i] = v;
}

__global__ void final_kernel(const float* __restrict__ h, const float* __restrict__ W,
                             const float* __restrict__ bp, float* __restrict__ out) {
    int b = blockIdx.x * 256 + threadIdx.x;
    if (b >= BB) return;
    float acc = bp[0];
    const float* hp = h + (size_t)b * 80;
#pragma unroll
    for (int k = 0; k < 80; k += 4) {
        float4 v = *(const float4*)(hp + k);
        float4 w = *(const float4*)(W + k);
        acc += v.x * w.x + v.y * w.y + v.z * w.z + v.w * w.w;
    }
    out[b] = sigmf(acc);
}

extern "C" void kernel_launch(void* const* d_in, const int* in_sizes, int n_in,
                              void* d_out, int out_size, void* d_ws, size_t ws_size,
                              hipStream_t stream) {
    (void)in_sizes; (void)n_in; (void)out_size; (void)ws_size;
    const float* user    = (const float*)d_in[0];
    const float* xhist   = (const float*)d_in[1];
    const float* item    = (const float*)d_in[2];
    const int*   mask    = (const int*)  d_in[3];
    const int*   slen    = (const int*)  d_in[4];
    const float* gru_Wih = (const float*)d_in[5];
    const float* gru_Whh = (const float*)d_in[6];
    const float* gru_bih = (const float*)d_in[7];
    const float* gru_bhh = (const float*)d_in[8];
    const float* aug_Wr  = (const float*)d_in[9];
    const float* aug_br  = (const float*)d_in[10];
    const float* aug_Wu  = (const float*)d_in[11];
    const float* aug_bu  = (const float*)d_in[12];
    const float* aug_Wh  = (const float*)d_in[13];
    const float* aug_bh  = (const float*)d_in[14];
    const float* att_W0  = (const float*)d_in[15];
    const float* att_b0  = (const float*)d_in[16];
    const float* att_W1  = (const float*)d_in[17];
    const float* att_b1  = (const float*)d_in[18];
    const float* att_W2  = (const float*)d_in[19];
    const float* att_b2  = (const float*)d_in[20];
    const float* ph_W0   = (const float*)d_in[21];
    const float* ph_b0   = (const float*)d_in[22];
    const float* ph_W1   = (const float*)d_in[23];
    const float* ph_b1   = (const float*)d_in[24];
    const float* ph_W2   = (const float*)d_in[25];
    const float* ph_b2   = (const float*)d_in[26];

    char* ws = (char*)d_ws;
    size_t off = 0;
    auto alloc = [&](size_t bytes) -> void* {
        void* p = (void*)(ws + off);
        off += (bytes + 255) & ~(size_t)255;
        return p;
    };
    const size_t MT = (size_t)BB * TT;
    float* GI      = (float*)alloc(MT * 384 * 4);
    float* keys    = (float*)alloc(MT * 128 * 4);
    bf16*  keys_bf = (bf16*) alloc(MT * 128 * 2);
    float* h1      = (float*)alloc(MT * 80 * 4);
    float* h2      = (float*)alloc(MT * 40 * 4);
    float* score   = (float*)alloc(MT * 4);
    float* attn    = (float*)alloc(MT * 4);
    float* pooled  = (float*)alloc((size_t)BB * 128 * 4);
    float* hist    = (float*)alloc((size_t)BB * 128 * 4);
    float* attf    = (float*)alloc((size_t)BB * 128 * 4);
    float* comb    = (float*)alloc((size_t)BB * 640 * 4);
    float* ph0o    = (float*)alloc((size_t)BB * 200 * 4);
    float* ph1o    = (float*)alloc((size_t)BB * 80 * 4);
    bf16* Wih_bf   = (bf16*)alloc(384 * 128 * 2);
    bf16* Whh_bf   = (bf16*)alloc(384 * 128 * 2);
    bf16* attW0_bf = (bf16*)alloc(80 * 512 * 2);
    bf16* attW1_bf = (bf16*)alloc(40 * 80 * 2);
    bf16* Wru_bf   = (bf16*)alloc(256 * 256 * 2);
    bf16* Wh_bf    = (bf16*)alloc(128 * 256 * 2);
    bf16* phW0_bf  = (bf16*)alloc(200 * 640 * 2);
    bf16* phW1_bf  = (bf16*)alloc(80 * 200 * 2);

    auto cvt = [&](const float* s, bf16* d, int n) {
        int n8 = n / 8;
        cvt_kernel<<<(n8 + 255) / 256, 256, 0, stream>>>(s, d, n8);
    };
    cvt(gru_Wih, Wih_bf, 384 * 128);
    cvt(gru_Whh, Whh_bf, 384 * 128);
    cvt(att_W0, attW0_bf, 80 * 512);
    cvt(att_W1, attW1_bf, 40 * 80);
    cvt(aug_Wr, Wru_bf, 128 * 256);                 // stacked [Wr; Wu]
    cvt(aug_Wu, Wru_bf + 128 * 256, 128 * 256);
    cvt(aug_Wh, Wh_bf, 128 * 256);
    cvt(ph_W0, phW0_bf, 200 * 640);
    cvt(ph_W1, phW1_bf, 80 * 200);

    hist_sum_kernel<<<BB, 128, 0, stream>>>(xhist, mask, slen, hist);

    // GI = X @ Wih^T + bih  (one big GEMM, removes input projection from the scan)
    gemm_nt_kernel<<<dim3((int)(MT / 64), 384 / 32), 256, 0, stream>>>(
        xhist, Wih_bf, gru_bih, GI, (int)MT, 384, 128, 0);

    gru_scan_kernel<<<BB / 16, 256, 0, stream>>>(GI, Whh_bf, gru_bhh, keys, keys_bf);

    att0_kernel<<<dim3((int)(MT / 64), 3), 256, 0, stream>>>(keys, item, attW0_bf, att_b0, h1);

    gemm_nt_kernel<<<dim3((int)(MT / 64), 2), 256, 0, stream>>>(
        h1, attW1_bf, att_b1, h2, (int)MT, 40, 80, 1);

    score_kernel<<<(int)((MT + 255) / 256), 256, 0, stream>>>(h2, att_W2, att_b2, score, (int)MT);
    softmax_kernel<<<BB, 128, 0, stream>>>(score, mask, attn);
    pooled_kernel<<<BB, 128, 0, stream>>>(attn, keys, pooled);

    augru_scan_kernel<<<BB / 16, 256, 0, stream>>>(
        keys_bf, Wru_bf, Wh_bf, aug_br, aug_bu, aug_bh, pooled, slen, attf);

    comb_kernel<<<(BB * 640 + 255) / 256, 256, 0, stream>>>(user, item, hist, attf, comb);

    gemm_nt_kernel<<<dim3(BB / 64, 7), 256, 0, stream>>>(comb, phW0_bf, ph_b0, ph0o, BB, 200, 640, 2);
    gemm_nt_kernel<<<dim3(BB / 64, 3), 256, 0, stream>>>(ph0o, phW1_bf, ph_b1, ph1o, BB, 80, 200, 2);

    final_kernel<<<(BB + 255) / 256, 256, 0, stream>>>(ph1o, ph_W2, ph_b2, (float*)d_out);
}
